// TransformerBlock_50783693308180
// MI455X (gfx1250) — matrix operand order repaired and checked by reading, not compiled, not run on hardware
//
#include <hip/hip_runtime.h>
#include <hip/hip_bf16.h>
#include <math.h>

// ---------------------------------------------------------------------------
// Transformer block for MI455X (gfx1250, wave32, WMMA).
// All matmul FLOPs mapped to v_wmma_f32_16x16x32_bf16 (bf16 in, f32 accum).
// ---------------------------------------------------------------------------

typedef __bf16 bf16;
typedef __attribute__((ext_vector_type(16))) __bf16 v16bf;
typedef __attribute__((ext_vector_type(8)))  __bf16 v8bf;
typedef __attribute__((ext_vector_type(4)))  __bf16 v4bf;
typedef __attribute__((ext_vector_type(8)))  float  v8f;

#define DMODEL 2048
#define FFDIM  8192
#define NTOK   4096   /* B*S = 2*2048 */
#define SEQ    2048
#define NHEAD  16
#define HDIM   128

static __device__ __forceinline__ v8f zero8() {
  v8f r;
#pragma unroll
  for (int i = 0; i < 8; ++i) r[i] = 0.0f;
  return r;
}

// A-fragment (16x32 bf16, MxK): lane<16 holds K {0..15}, lane>=16 holds
// K {16..31}. p points at (row, k + half*16); p+8 covers the upper 8.
static __device__ __forceinline__ v16bf load_a16(const bf16* p) {
  v8bf lo = *(const v8bf*)p;
  v8bf hi = *(const v8bf*)(p + 8);
  v16bf r;
#pragma unroll
  for (int i = 0; i < 8; ++i) { r[i] = lo[i]; r[i + 8] = hi[i]; }
  return r;
}

static __device__ __forceinline__ v8f wmma_bf16(v16bf a, v16bf b, v8f c) {
  return __builtin_amdgcn_wmma_f32_16x16x32_bf16(
      /*neg_a=*/false, a, /*neg_b=*/false, b,
      /*c_mod=*/(short)0, c, /*reuse_a=*/false, /*reuse_b=*/false);
}

static __device__ __forceinline__ float gelu_f(float x) {
  return 0.5f * x * (1.0f + erff(x * 0.7071067811865476f));
}

// ---------------------------------------------------------------------------
// fp32 -> bf16 weight cast (weights then live as 96MB bf16, resident in L2)
// ---------------------------------------------------------------------------
__global__ __launch_bounds__(256) void cvt_kernel(const float* __restrict__ in,
                                                  bf16* __restrict__ out,
                                                  int n4) {
  int i = blockIdx.x * 256 + threadIdx.x;
  if (i >= n4) return;
  float4 f = ((const float4*)in)[i];
  v4bf o;
  o[0] = (bf16)f.x; o[1] = (bf16)f.y; o[2] = (bf16)f.z; o[3] = (bf16)f.w;
  *(v4bf*)(out + (size_t)i * 4) = o;
}

// ---------------------------------------------------------------------------
// LayerNorm: one 256-thread block per row of 2048; bf16 output
// ---------------------------------------------------------------------------
__global__ __launch_bounds__(256) void ln_kernel(const float* __restrict__ x,
                                                 const float* __restrict__ gamma,
                                                 const float* __restrict__ beta,
                                                 bf16* __restrict__ out) {
  __shared__ float s1[256];
  __shared__ float s2[256];
  const int tid = threadIdx.x;
  const size_t row = blockIdx.x;
  const float* xr = x + row * DMODEL;
  float v[8], sum = 0.f, sq = 0.f;
#pragma unroll
  for (int j = 0; j < 8; ++j) {
    v[j] = xr[tid + j * 256];
    sum += v[j];
    sq  += v[j] * v[j];
  }
  s1[tid] = sum; s2[tid] = sq;
  __syncthreads();
  for (int off = 128; off > 0; off >>= 1) {
    if (tid < off) { s1[tid] += s1[tid + off]; s2[tid] += s2[tid + off]; }
    __syncthreads();
  }
  const float mu   = s1[0] * (1.0f / DMODEL);
  const float var  = s2[0] * (1.0f / DMODEL) - mu * mu;
  const float rinv = rsqrtf(var + 1e-5f);
  bf16* orow = out + row * DMODEL;
#pragma unroll
  for (int j = 0; j < 8; ++j) {
    int i = tid + j * 256;
    orow[i] = (bf16)((v[j] - mu) * rinv * gamma[i] + beta[i]);
  }
}

// ---------------------------------------------------------------------------
// Generic bf16 GEMM: C[M,N] = A[M,K] @ W[N,K]^T (+bias), fused epilogues.
// Block: 256 threads = 8 waves, block tile 64(M) x 128(N), wave tile 32x32.
// ---------------------------------------------------------------------------
#define EPI_B16   0   // store bf16
#define EPI_B16T  1   // store bf16 transposed (Vt: [feature, token])
#define EPI_GELU  2   // gelu -> bf16
#define EPI_RES   3   // fp32 out = res + acc + bias
#define EPI_ADD   4   // fp32 out += acc + bias  (in-place residual)

template <int EPI>
__global__ __launch_bounds__(256)
void gemm_kernel(const bf16* __restrict__ A, const bf16* __restrict__ W,
                 const float* __restrict__ bias, const float* __restrict__ res,
                 float* __restrict__ outf, bf16* __restrict__ outb,
                 int M, int N, int K) {
  const int tid  = threadIdx.x;
  const int lane = tid & 31;
  const int w    = tid >> 5;
  const int lrow = lane & 15;
  const int half = lane >> 4;
  const int m_wave = blockIdx.y * 64  + (w >> 2) * 32;
  const int n_wave = blockIdx.x * 128 + (w & 3) * 32;

  v8f acc[2][2];
#pragma unroll
  for (int i = 0; i < 2; ++i)
#pragma unroll
    for (int j = 0; j < 2; ++j) acc[i][j] = zero8();

  const bf16* a0p = A + (size_t)(m_wave + lrow) * K + half * 16;
  const bf16* a1p = a0p + (size_t)16 * K;
  const bf16* b0p = W + (size_t)(n_wave + lrow) * K + half * 16;
  const bf16* b1p = b0p + (size_t)16 * K;

  for (int k = 0; k < K; k += 32) {
    __builtin_prefetch(b0p + k + 256, 0, 0);   // global_prefetch next W block
    __builtin_prefetch(b1p + k + 256, 0, 0);
    v16bf a0 = load_a16(a0p + k);
    v16bf a1 = load_a16(a1p + k);
    v16bf b0 = *(const v16bf*)(b0p + k);       // contiguous 32B B-fragment
    v16bf b1 = *(const v16bf*)(b1p + k);
    acc[0][0] = wmma_bf16(a0, b0, acc[0][0]);
    acc[0][1] = wmma_bf16(a0, b1, acc[0][1]);
    acc[1][0] = wmma_bf16(a1, b0, acc[1][0]);
    acc[1][1] = wmma_bf16(a1, b1, acc[1][1]);
  }

#pragma unroll
  for (int tm = 0; tm < 2; ++tm)
#pragma unroll
    for (int tn = 0; tn < 2; ++tn) {
      const int m0 = m_wave + tm * 16 + half * 8;   // C layout: lane-half row offset
      const int n  = n_wave + tn * 16 + lrow;
      const float bv = bias[n];
      if (EPI == EPI_B16T) {
        v8bf pk;
#pragma unroll
        for (int r = 0; r < 8; ++r) pk[r] = (bf16)(acc[tm][tn][r] + bv);
        *(v8bf*)(outb + (size_t)n * M + m0) = pk;   // rows contiguous in token dim
      } else {
#pragma unroll
        for (int r = 0; r < 8; ++r) {
          const size_t idx = (size_t)(m0 + r) * N + n;
          const float vv = acc[tm][tn][r] + bv;
          if (EPI == EPI_B16)  outb[idx] = (bf16)vv;
          if (EPI == EPI_GELU) outb[idx] = (bf16)gelu_f(vv);
          if (EPI == EPI_RES)  outf[idx] = res[idx] + vv;
          if (EPI == EPI_ADD)  outf[idx] = outf[idx] + vv;
        }
      }
    }
}

// ---------------------------------------------------------------------------
// Flash attention: one wave per (batch, head, 16-row q tile). Causal.
// Q,K: bf16 [token, 2048]; Vt: bf16 [feature, token]; ctx: bf16 [token, 2048]
// ---------------------------------------------------------------------------
#define ATT_WAVES 4
__global__ __launch_bounds__(32 * ATT_WAVES)
void attn_kernel(const bf16* __restrict__ Q, const bf16* __restrict__ Kmat,
                 const bf16* __restrict__ Vt, bf16* __restrict__ ctx) {
  __shared__ __align__(16) bf16 lp[ATT_WAVES][16][40];  // P tile staging, padded

  const int tid  = threadIdx.x;
  const int lane = tid & 31;
  const int w    = tid >> 5;
  const int wg   = blockIdx.x * ATT_WAVES + w;
  const int qt   = wg & 127;           // q tile within head (S/16 = 128)
  const int hh   = (wg >> 7) & 15;     // head
  const int b    = wg >> 11;           // batch
  const int lrow = lane & 15;
  const int half = lane >> 4;

  // Q fragments: A-matrix 16 x 128, 4 chunks of K=32
  const bf16* qp = Q + (size_t)(b * SEQ + qt * 16 + lrow) * DMODEL + hh * HDIM + half * 16;
  v16bf qf[4];
#pragma unroll
  for (int c = 0; c < 4; ++c) qf[c] = load_a16(qp + c * 32);

  float mr[8], lr[8];
  v8f o[8];
#pragma unroll
  for (int r = 0; r < 8; ++r) { mr[r] = -1e30f; lr[r] = 0.0f; }
#pragma unroll
  for (int nt = 0; nt < 8; ++nt) o[nt] = zero8();

  const float scale = 0.08838834764831845f;  // 1/sqrt(128)
  const int kend = qt * 16 + 16;             // causal limit (exclusive)

  for (int kb = 0; kb < kend; kb += 32) {
    // ---- scores S = Q K^T : two 16x16 tiles (tokens kb..+15, kb+16..+31)
    v8f s0 = zero8(), s1 = zero8();
#pragma unroll
    for (int c = 0; c < 4; ++c) {
      const bf16* kp = Kmat + (size_t)(b * SEQ + kb + lrow) * DMODEL +
                       hh * HDIM + c * 32 + half * 16;
      v16bf k0 = *(const v16bf*)kp;
      v16bf k1 = *(const v16bf*)(kp + (size_t)16 * DMODEL);
      s0 = wmma_bf16(qf[c], k0, s0);
      s1 = wmma_bf16(qf[c], k1, s1);
    }

    // ---- online softmax (rows 0..7 in lanes 0-15, rows 8..15 in lanes 16-31)
    const int n0 = kb + lrow, n1 = n0 + 16;
#pragma unroll
    for (int r = 0; r < 8; ++r) {
      const int qrow = qt * 16 + half * 8 + r;
      float e0 = s0[r] * scale; if (n0 > qrow) e0 = -1e30f;
      float e1 = s1[r] * scale; if (n1 > qrow) e1 = -1e30f;
      float t = fmaxf(e0, e1);
#pragma unroll
      for (int off = 1; off < 16; off <<= 1) t = fmaxf(t, __shfl_xor(t, off, 32));
      const float mnew  = fmaxf(mr[r], t);
      const float alpha = __expf(mr[r] - mnew);
      const float p0 = __expf(e0 - mnew);
      const float p1 = __expf(e1 - mnew);
      float rs = p0 + p1;
#pragma unroll
      for (int off = 1; off < 16; off <<= 1) rs += __shfl_xor(rs, off, 32);
      lr[r] = lr[r] * alpha + rs;
      mr[r] = mnew;
#pragma unroll
      for (int nt = 0; nt < 8; ++nt) o[nt][r] *= alpha;
      const int prow = half * 8 + r;
      lp[w][prow][lrow]      = (bf16)p0;   // C-layout -> LDS
      lp[w][prow][16 + lrow] = (bf16)p1;
    }

    // Same-wave DS ops retire in order; fence stops compiler reordering.
    asm volatile("s_wait_dscnt 0" ::: "memory");

    // ---- reload P as A-fragment (16 x 32) and accumulate O += P V
    v16bf pf = load_a16(&lp[w][lrow][half * 16]);
#pragma unroll
    for (int nt = 0; nt < 8; ++nt) {
      const bf16* vp = Vt + (size_t)(hh * HDIM + nt * 16 + lrow) * NTOK +
                       b * SEQ + kb + half * 16;
      v16bf vf = *(const v16bf*)vp;       // contiguous over tokens (V transposed)
      o[nt] = wmma_bf16(pf, vf, o[nt]);
    }
  }

  // ---- finalize: ctx = O / l, store bf16 [token, D]
#pragma unroll
  for (int r = 0; r < 8; ++r) {
    const float inv = 1.0f / lr[r];
    bf16* cp = ctx + (size_t)(b * SEQ + qt * 16 + half * 8 + r) * DMODEL +
               hh * HDIM + lrow;
#pragma unroll
    for (int nt = 0; nt < 8; ++nt) cp[nt * 16] = (bf16)(o[nt][r] * inv);
  }
}

// ---------------------------------------------------------------------------
// Host launch
// ---------------------------------------------------------------------------
extern "C" void kernel_launch(void* const* d_in, const int* in_sizes, int n_in,
                              void* d_out, int out_size, void* d_ws, size_t ws_size,
                              hipStream_t stream) {
  (void)in_sizes; (void)n_in; (void)out_size; (void)ws_size;
  const float* x   = (const float*)d_in[0];
  const float* qw  = (const float*)d_in[1];
  const float* kw  = (const float*)d_in[2];
  const float* vw  = (const float*)d_in[3];
  const float* ow  = (const float*)d_in[4];
  const float* qbi = (const float*)d_in[5];
  const float* kbi = (const float*)d_in[6];
  const float* vbi = (const float*)d_in[7];
  const float* obi = (const float*)d_in[8];
  const float* w1  = (const float*)d_in[9];
  const float* w2  = (const float*)d_in[10];
  const float* b1  = (const float*)d_in[11];
  const float* b2  = (const float*)d_in[12];
  const float* g1  = (const float*)d_in[13];
  const float* be1 = (const float*)d_in[14];
  const float* g2  = (const float*)d_in[15];
  const float* be2 = (const float*)d_in[16];
  float* out = (float*)d_out;

  char* ws = (char*)d_ws;
  const size_t MB = 1024ull * 1024ull;
  bf16* h   = (bf16*)(ws +   0 * MB);   // ln output (reused for ln2)
  bf16* Qb  = (bf16*)(ws +  16 * MB);
  bf16* Kb  = (bf16*)(ws +  32 * MB);
  bf16* Vt  = (bf16*)(ws +  48 * MB);   // [feature, token]
  bf16* ctx = (bf16*)(ws +  64 * MB);
  bf16* f   = (bf16*)(ws +  80 * MB);   // [4096, 8192]
  bf16* wqb = (bf16*)(ws + 144 * MB);
  bf16* wkb = (bf16*)(ws + 152 * MB);
  bf16* wvb = (bf16*)(ws + 160 * MB);
  bf16* wob = (bf16*)(ws + 168 * MB);
  bf16* w1b = (bf16*)(ws + 176 * MB);
  bf16* w2b = (bf16*)(ws + 208 * MB);

  // Cast weights to bf16 (96MB total -> L2 resident)
  const int nDD4 = (DMODEL * DMODEL) / 4;   // 1,048,576
  const int nFD4 = (FFDIM * DMODEL) / 4;    // 4,194,304
  cvt_kernel<<<nDD4 / 256, 256, 0, stream>>>(qw, wqb, nDD4);
  cvt_kernel<<<nDD4 / 256, 256, 0, stream>>>(kw, wkb, nDD4);
  cvt_kernel<<<nDD4 / 256, 256, 0, stream>>>(vw, wvb, nDD4);
  cvt_kernel<<<nDD4 / 256, 256, 0, stream>>>(ow, wob, nDD4);
  cvt_kernel<<<nFD4 / 256, 256, 0, stream>>>(w1, w1b, nFD4);
  cvt_kernel<<<nFD4 / 256, 256, 0, stream>>>(w2, w2b, nFD4);

  // LN1
  ln_kernel<<<NTOK, 256, 0, stream>>>(x, g1, be1, h);

  // QKV projections
  dim3 gD(DMODEL / 128, NTOK / 64);  // (16, 64)
  gemm_kernel<EPI_B16 ><<<gD, 256, 0, stream>>>(h, wqb, qbi, nullptr, nullptr, Qb,
                                                NTOK, DMODEL, DMODEL);
  gemm_kernel<EPI_B16 ><<<gD, 256, 0, stream>>>(h, wkb, kbi, nullptr, nullptr, Kb,
                                                NTOK, DMODEL, DMODEL);
  gemm_kernel<EPI_B16T><<<gD, 256, 0, stream>>>(h, wvb, vbi, nullptr, nullptr, Vt,
                                                NTOK, DMODEL, DMODEL);

  // Causal flash attention: 2*16*128 = 4096 waves, 4 per block
  attn_kernel<<<4096 / ATT_WAVES, 32 * ATT_WAVES, 0, stream>>>(Qb, Kb, Vt, ctx);

  // O-projection + residual -> d_out (fp32)
  gemm_kernel<EPI_RES><<<gD, 256, 0, stream>>>(ctx, wob, obi, x, out, nullptr,
                                               NTOK, DMODEL, DMODEL);

  // LN2 on (x + attn_out)
  ln_kernel<<<NTOK, 256, 0, stream>>>(out, g2, be2, h);

  // FFN1 with exact GELU -> bf16 f
  dim3 gF1(FFDIM / 128, NTOK / 64);  // (64, 64)
  gemm_kernel<EPI_GELU><<<gF1, 256, 0, stream>>>(h, w1b, b1, nullptr, nullptr, f,
                                                 NTOK, FFDIM, DMODEL);

  // FFN2 + in-place residual add into d_out
  gemm_kernel<EPI_ADD><<<gD, 256, 0, stream>>>(f, w2b, b2, nullptr, out, nullptr,
                                               NTOK, DMODEL, FFDIM);
}
